// STFT_60103772340263
// MI455X (gfx1250) — matrix operand order repaired and checked by reading, not compiled, not run on hardware
//
#include <hip/hip_runtime.h>

typedef __attribute__((ext_vector_type(16))) __bf16 v16bf;
typedef __attribute__((ext_vector_type(8)))  float  v8f;

union AU { v16bf v; uint4 q[2]; };

#define NFFT      256
#define HOP       128
#define CUTOFF    129
#define NSAMP     256000
#define NFRAMES   1999
#define NBATCH    64
#define MROWS     320      // 258 interleaved (re/im) rows padded to 20*16 (zeros)
#define NTILES    125      // ceil(1999/16)
#define KSTEPS    8        // 256 / 32
#define NWAVES    5        // waves per block
#define JTILES    4        // tiles per wave: m = wavei + 5*j, j = 0..3
#define TSTRIDE   (NWAVES * 16 * NFFT)   // 20480 elements between a wave's tiles

#define TWO_PI_F 6.28318530717958647692f

// ---------------- basis preparation: bf16, interleaved re/im rows ----------
// row 2k   =  cos(2*pi*k*n/256) * hann(n)
// row 2k+1 = -sin(2*pi*k*n/256) * hann(n)
// rows 258..319 = 0 (pad tiles; only k=128 of tile 16 is stored)
__global__ void stft_basis_kernel(__bf16* __restrict__ basis) {
    const int r = blockIdx.x;     // 0..319
    const int n = threadIdx.x;    // 0..255
    float val = 0.0f;
    if (r < 2 * CUTOFF) {
        const int k = r >> 1;
        const int kn = (k * n) & (NFFT - 1);   // exact integer phase reduction
        const float ph = (TWO_PI_F / (float)NFFT) * (float)kn;
        const float w  = 0.5f * (1.0f - __cosf((TWO_PI_F / (float)NFFT) * (float)n));
        val = ((r & 1) ? -__sinf(ph) : __cosf(ph)) * w;
    }
    basis[r * NFFT + n] = (__bf16)val;
}

// f32x16 -> bf16x16 as this kernel's B operand (16 consecutive K values, same k order as A)
static __device__ __forceinline__ v16bf cvt16(const float4 f0, const float4 f1,
                                              const float4 f2, const float4 f3) {
    v16bf bm;
    bm[0]  = (__bf16)f0.x; bm[1]  = (__bf16)f0.y; bm[2]  = (__bf16)f0.z; bm[3]  = (__bf16)f0.w;
    bm[4]  = (__bf16)f1.x; bm[5]  = (__bf16)f1.y; bm[6]  = (__bf16)f1.z; bm[7]  = (__bf16)f1.w;
    bm[8]  = (__bf16)f2.x; bm[9]  = (__bf16)f2.y; bm[10] = (__bf16)f2.z; bm[11] = (__bf16)f2.w;
    bm[12] = (__bf16)f3.x; bm[13] = (__bf16)f3.y; bm[14] = (__bf16)f3.z; bm[15] = (__bf16)f3.w;
    return bm;
}

// magnitude + 4 stores for one C tile (caller guarantees k-range < 129)
static __device__ __forceinline__ void store_tile(const v8f c, float* __restrict__ ob, const int k0) {
#pragma unroll
    for (int p = 0; p < 4; ++p) {
        const float re = c[2 * p];
        const float im = c[2 * p + 1];
        ob[(size_t)(k0 + p) * NFRAMES] = __builtin_sqrtf(fmaxf(re * re + im * im, 1e-16f));
    }
}

// ---------------- WMMA STFT GEMM + magnitude ------------------------------
// grid: (NTILES, NBATCH), block: 160 threads = 5 waves.
// Wave w (uniform via readfirstlane) owns tiles {w, w+5, w+10, w+15}: exactly
// 4 double-buffered tiles per wave -> branch-free, phi-free hot loop with
// EXEC all-1s by construction; tiles 17..19 hit the zero pad rows.
__global__ __launch_bounds__(160) void stft_wmma_kernel(
    const float* __restrict__ x,
    const __bf16* __restrict__ basis,
    float* __restrict__ out)
{
    const int nt    = blockIdx.x;             // N-tile (frames)
    const int b     = blockIdx.y;             // batch
    const int wavei = __builtin_amdgcn_readfirstlane((int)(threadIdx.x >> 5)); // SGPR-uniform
    const int lane  = (int)(threadIdx.x & 31);
    const int half  = lane >> 4;              // lane group 0/1
    const int l16   = lane & 15;

    // B-operand column for this lane: frame t; K-halves split by lane group
    const int t  = nt * 16 + l16;
    const int tc = (t < NFRAMES) ? t : (NFRAMES - 1);   // clamp addr, guard store
    const float* xcol = x + (size_t)b * NSAMP + (size_t)tc * HOP + half * 16;

    // A-operand per-lane base pointer; all tile/k-step displacements are
    // compile-time constants that fold into load immediate offsets.
    const __bf16* aw = basis + (size_t)l16 * NFFT + half * 16 + (size_t)wavei * (16 * NFFT);

    AU     a[2][JTILES];
    float4 f[2][4];

    // ---- prologue: k-step 0 operand loads
#pragma unroll
    for (int j = 0; j < JTILES; ++j) {
        a[0][j].q[0] = *(const uint4*)(aw + j * TSTRIDE);
        a[0][j].q[1] = *(const uint4*)(aw + j * TSTRIDE + 8);
    }
    f[0][0] = *(const float4*)(xcol + 0);
    f[0][1] = *(const float4*)(xcol + 4);
    f[0][2] = *(const float4*)(xcol + 8);
    f[0][3] = *(const float4*)(xcol + 12);

    v8f acc[JTILES];
    const v8f vz = {0.f, 0.f, 0.f, 0.f, 0.f, 0.f, 0.f, 0.f};
#pragma unroll
    for (int j = 0; j < JTILES; ++j) acc[j] = vz;

#pragma unroll
    for (int ks = 0; ks < KSTEPS; ++ks) {
        const int cur = ks & 1;
        const int nxt = cur ^ 1;

        // ---- prefetch next k-step operands (compile-time guard, no branch)
        if (ks < KSTEPS - 1) {
            const int o = (ks + 1) * 32;
#pragma unroll
            for (int j = 0; j < JTILES; ++j) {
                a[nxt][j].q[0] = *(const uint4*)(aw + j * TSTRIDE + o);
                a[nxt][j].q[1] = *(const uint4*)(aw + j * TSTRIDE + o + 8);
            }
            f[nxt][0] = *(const float4*)(xcol + o + 0);
            f[nxt][1] = *(const float4*)(xcol + o + 4);
            f[nxt][2] = *(const float4*)(xcol + o + 8);
            f[nxt][3] = *(const float4*)(xcol + o + 12);
        }

        // ---- consume current buffers: 4 independent WMMA chains, no branches
        const v16bf bm = cvt16(f[cur][0], f[cur][1], f[cur][2], f[cur][3]);
#pragma unroll
        for (int j = 0; j < JTILES; ++j) {
            acc[j] = __builtin_amdgcn_wmma_f32_16x16x32_bf16(
                false, a[cur][j].v, false, bm, (short)0, acc[j], false, false);
        }
    }

    // ---- magnitude + stores (re/im pairs live in adjacent accumulator VGPRs)
    // k = (wavei + 5*j)*8 + half*4 + p
    if (t < NFRAMES) {
        float* ob = out + (size_t)b * CUTOFF * NFRAMES + t;
        const int k0 = wavei * 8 + half * 4;
        store_tile(acc[0], ob, k0 + 0 * 40);          // m <= 14 -> k <= 119
        store_tile(acc[1], ob, k0 + 1 * 40);
        store_tile(acc[2], ob, k0 + 2 * 40);
        if (wavei == 0) {                             // m = 15 -> k = 120..127
            store_tile(acc[3], ob, k0 + 3 * 40);
        }
        if (wavei == 1 && half == 0) {                // m = 16 -> only k = 128 real
            const float re = acc[3][0];
            const float im = acc[3][1];
            ob[(size_t)128 * NFRAMES] = __builtin_sqrtf(fmaxf(re * re + im * im, 1e-16f));
        }
        // waves 2..4, j=3 (m = 17..19): zero pad rows, nothing stored
    }
}

extern "C" void kernel_launch(void* const* d_in, const int* in_sizes, int n_in,
                              void* d_out, int out_size, void* d_ws, size_t ws_size,
                              hipStream_t stream) {
    const float* x     = (const float*)d_in[0];
    float*       out   = (float*)d_out;
    __bf16*      basis = (__bf16*)d_ws;     // 320*256*2 = 163840 bytes of scratch

    (void)in_sizes; (void)n_in; (void)out_size; (void)ws_size;

    // 1) build bf16 interleaved DFT basis (stream-ordered before the GEMM)
    stft_basis_kernel<<<dim3(MROWS), dim3(NFFT), 0, stream>>>(basis);

    // 2) WMMA STFT: one block per (16-frame tile, batch)
    stft_wmma_kernel<<<dim3(NTILES, NBATCH), dim3(NWAVES * 32), 0, stream>>>(x, basis, out);
}
